// KPNet_50680614093404
// MI455X (gfx1250) — matrix operand order repaired and checked by reading, not compiled, not run on hardware
//
#include <hip/hip_runtime.h>
#include <hip/hip_bf16.h>
#include <math.h>

typedef __attribute__((ext_vector_type(16))) _Float16 v16h;
typedef __attribute__((ext_vector_type(8)))  _Float16 v8h;
typedef __attribute__((ext_vector_type(8)))  float    v8f;

#define BB   4
#define NN   16384
#define SS   2048
#define KK   32
#define KD   64            // K*DELTA
#define C1D  64
#define C2D  128
#define C3D  256
#define NQ   (BB * SS)     // 8192
#define INV_BN 0.99999500003749968f   // 1/sqrt(1+1e-5)

#define WMMA16(a, b, c) __builtin_amdgcn_wmma_f32_16x16x32_f16(false, a, false, b, (short)0, c, false, false)

// ---------------------------------------------------------------------------
// Fragment-swizzled weight staging with BN scale folded into the weights:
// scale[m]*(W x) == (scale*W) x.  A-fragment layout (contiguous k order,
// matching B): lane<16: row=lane, halves0-15=K0..15; lane>=16:
// row=lane-16, K16..31.  LDS holds fragment(mt,kk) for lane L as 16
// contiguous halves at ((mt*tilesK+kk)*512 + L*16) -> one 32B read per lane.
// ---------------------------------------------------------------------------
__device__ __forceinline__ void swizzle_weight_scaled(const float* __restrict__ Wf,
                                                      const float* __restrict__ gamma,
                                                      _Float16* __restrict__ Wl,
                                                      int M, int Kd, int tid, int nthr) {
  int tilesK = Kd >> 5;
  for (int i = tid; i < M * Kd; i += nthr) {
    int row = i / Kd, k = i % Kd;            // global is row-major [M][Kd]
    int kk = k >> 5, kq = k & 31;
    int L  = (row & 15) | (((kq >> 4) & 1) << 4);
    int h  = kq & 15;
    int mt = row >> 4;
    float sc = gamma[row] * INV_BN;
    Wl[(((mt * tilesK + kk) << 9) + (L << 4) + h)] = (_Float16)(Wf[i] * sc);
  }
}

__device__ __forceinline__ v16h load_frag_a_sw(const _Float16* Wl, int tilesK,
                                               int mt, int kk, int lane) {
  return *(const v16h*)(Wl + (((mt * tilesK + kk) << 9) + (lane << 4)));
}

// B activations stored column-major [32 cols][Kd]: 16 contiguous halves/lane.
__device__ __forceinline__ v16h load_frag_b_cm(const _Float16* In, int Kd,
                                               int k0, int n0, int lane) {
  int col = n0 + (lane & 15);
  int ks  = k0 + ((lane >> 4) << 4);
  return *(const v16h*)(In + col * Kd + ks);
}

// One wave MLP layer:  D[M x 32] = relu((scaledW*In) + cb).
// Dual accumulation chains (both n-subtiles) share each A-fragment: keeps
// register pressure low (no spills in the hot query loop) while overlapping
// the WMMA->VALU hazard window with the second chain.
// In: col-major [32][Kd] f16.  outH: col-major [32][M] f16.  outF: [32][M] f32.
__device__ __forceinline__ void wave_layer_wmma(const _Float16* Wl, int M, int Kd,
                                                const float* cb,
                                                const _Float16* In,
                                                _Float16* outH, float* outF, int lane) {
  int tilesK = Kd >> 5;
  for (int mt = 0; mt < (M >> 4); ++mt) {
    v8f acc0 = {}, acc1 = {};
    for (int kk = 0; kk < tilesK; ++kk) {
      v16h a  = load_frag_a_sw(Wl, tilesK, mt, kk, lane);
      v16h b0 = load_frag_b_cm(In, Kd, kk * 32, 0,  lane);
      v16h b1 = load_frag_b_cm(In, Kd, kk * 32, 16, lane);
      acc0 = WMMA16(a, b0, acc0);
      acc1 = WMMA16(a, b1, acc1);
    }
    int col0  = (lane & 15);
    int rbase = mt * 16 + ((lane >> 4) << 3);
    v8f cbv = *(const v8f*)(cb + rbase);       // 8 contiguous channel consts
    v8h oh0, oh1; v8f of0, of1;
#pragma unroll
    for (int r = 0; r < 8; ++r) {
      float v0 = acc0[r] + cbv[r];
      float v1 = acc1[r] + cbv[r];
      v0 = v0 > 0.f ? v0 : 0.f;
      v1 = v1 > 0.f ? v1 : 0.f;
      oh0[r] = (_Float16)v0; of0[r] = v0;
      oh1[r] = (_Float16)v1; of1[r] = v1;
    }
    if (outH) {
      *(v8h*)(outH + col0 * M + rbase)        = oh0;
      *(v8h*)(outH + (col0 + 16) * M + rbase) = oh1;
    }
    if (outF) {
      *(v8f*)(outF + col0 * M + rbase)        = of0;
      *(v8f*)(outF + (col0 + 16) * M + rbase) = of1;
    }
  }
}

// ---------------------------------------------------------------------------
// Kernel A: exact top-64 NN per query, then dilation-select 32 indices.
// d2 stored as order-preserving u32 keys; per-thread local minima cached in
// LDS.  Each round: 256->1 u64 argmin + single-owner 64-key rescan (only the
// owner's local min can change after poisoning the winner).  Key low bits are
// the index -> ties break to the lower index, matching stable top_k.
// ---------------------------------------------------------------------------
__global__ void knn_topk(const float* __restrict__ pc,
                         const int* __restrict__ randIdx,
                         const int* __restrict__ dil,
                         int* __restrict__ idxOut) {
  extern __shared__ unsigned char smemA[];
  unsigned int* sd = (unsigned int*)smemA;                   // 16384 keys
  uint4* sd4 = (uint4*)smemA;
  unsigned long long* tmin = (unsigned long long*)(sd + NN); // 256
  unsigned long long* winb = tmin + 256;                     // 1
  float* q = (float*)(winb + 1);                             // 4
  int* sel = (int*)(q + 4);                                  // 64

  int bq = blockIdx.x;                   // 0..NQ-1
  int b  = bq / SS, s = bq % SS;
  int tid = threadIdx.x;

  if (tid == 0) {
    int qi = randIdx[s];
    const float* qp = pc + ((size_t)b * NN + qi) * 7;
    q[0] = qp[0]; q[1] = qp[1]; q[2] = qp[2];
    q[3] = qp[0]*qp[0] + qp[1]*qp[1] + qp[2]*qp[2];
  }
  __syncthreads();
  float qx = q[0], qy = q[1], qz = q[2], qn = q[3];

  // fill keys (thread owns 4-key groups strided by 1024) + local min
  unsigned long long bk = ~0ull;
  for (int j = 0; j < 16; ++j) {
    int nbase = tid * 4 + j * 1024;
    uint4 keys;
#pragma unroll
    for (int e = 0; e < 4; ++e) {
      int n = nbase + e;
      const float* p = pc + ((size_t)b * NN + n) * 7;
      float px = p[0], py = p[1], pz = p[2];
      float pn = px*px + py*py + pz*pz;
      float d2 = qn + pn - 2.0f * (qx*px + qy*py + qz*pz);
      unsigned int u = __float_as_uint(d2);
      unsigned int mk = u ^ ((u & 0x80000000u) ? 0xFFFFFFFFu : 0x80000000u);
      (&keys.x)[e] = mk;
      unsigned long long key = ((unsigned long long)mk << 32) | (unsigned)n;
      bk = key < bk ? key : bk;
    }
    sd4[tid + j * 256] = keys;
  }
  tmin[tid] = bk;
  __syncthreads();

  for (int r = 0; r < KD; ++r) {
    if (tid < 32) {
      unsigned long long fk = tmin[tid];
#pragma unroll
      for (int w = 1; w < 8; ++w) {
        unsigned long long t = tmin[tid + w * 32];
        fk = t < fk ? t : fk;
      }
      for (int off = 16; off > 0; off >>= 1) {
        unsigned long long ok = __shfl_xor(fk, off, 32);
        fk = ok < fk ? ok : fk;
      }
      if (tid == 0) { winb[0] = fk; sel[r] = (int)(fk & 0xFFFFFFFFull); }
    }
    __syncthreads();
    int n = (int)(winb[0] & 0xFFFFFFFFull);
    int owner = (n >> 2) & 255;
    if (tid == owner) {
      sd[n] = 0xFFFFFFFFu;                 // poison winner
      unsigned long long nb = ~0ull;
      for (int j = 0; j < 16; ++j) {
        uint4 keys = sd4[tid + j * 256];
        int nbase = tid * 4 + j * 1024;
#pragma unroll
        for (int e = 0; e < 4; ++e) {
          unsigned long long key =
              ((unsigned long long)(&keys.x)[e] << 32) | (unsigned)(nbase + e);
          nb = key < nb ? key : nb;
        }
      }
      tmin[tid] = nb;
    }
    __syncthreads();
  }
  if (tid < KK) idxOut[(size_t)bq * KK + tid] = sel[dil[tid]];
}

// ---------------------------------------------------------------------------
// Kernel B: gather neighbors, layer1 (VALU f32), layers 2-3 (WMMA f16/f32acc),
// attention softmax, keypoints, pooled gfeat.  One wave per query; 4 waves /
// block share scale-folded fragment-swizzled f16 weights.
// ---------------------------------------------------------------------------
#define PER_WAVE_BYTES (4096 + 8192 + 32768 + 128)   // h1, h2, emb(f32), attn
__global__ void cluster_mlp(const float* __restrict__ pc,
                            const int* __restrict__ randIdx,
                            const int* __restrict__ idxIn,
                            const float* __restrict__ w1,
                            const float* __restrict__ gamma1, const float* __restrict__ beta1,
                            const float* __restrict__ w2,
                            const float* __restrict__ gamma2, const float* __restrict__ beta2,
                            const float* __restrict__ w3,
                            const float* __restrict__ gamma3, const float* __restrict__ beta3,
                            float* __restrict__ kpOut,       // (B,3,S)
                            float* __restrict__ gfeatOut) {  // [NQ][256] f32
  extern __shared__ unsigned char smemB[];
  float* w1f = (float*)smemB;                  // 512 (scale-folded)
  float* be1 = w1f + 512;                      // 64
  float* cb2 = be1 + 64;                       // 128
  float* cb3 = cb2 + 128;                      // 256   -> 960 floats total
  _Float16* w2h = (_Float16*)(cb3 + 256);      // 8192 halves (swizzled, scaled)
  _Float16* w3h = w2h + 8192;                  // 32768 halves (swizzled, scaled)
  unsigned char* pw = (unsigned char*)(w3h + 32768);

  int tid = threadIdx.x, lane = tid & 31, wv = tid >> 5;

  for (int i = tid; i < 512; i += 128) w1f[i] = w1[i] * gamma1[i >> 3] * INV_BN;
  for (int i = tid; i < 64;  i += 128) be1[i] = beta1[i];
  for (int i = tid; i < 128; i += 128) cb2[i] = beta2[i];
  for (int i = tid; i < 256; i += 128) cb3[i] = beta3[i];
  swizzle_weight_scaled(w2, gamma2, w2h, C2D, C1D, tid, 128);
  swizzle_weight_scaled(w3, gamma3, w3h, C3D, C2D, tid, 128);
  __syncthreads();

  unsigned char* mybase = pw + (size_t)wv * PER_WAVE_BYTES;
  _Float16* h1h  = (_Float16*)mybase;          // [32][64]  col-major
  _Float16* h2h  = h1h + 2048;                 // [32][128] col-major
  float*    embf = (float*)(h2h + 4096);       // [32][256] col-major f32
  float*    attl = embf + 32 * C3D;            // 32

  int nWaves = gridDim.x * 4;
  for (int qidx = blockIdx.x * 4 + wv; qidx < NQ; qidx += nWaves) {
    int b = qidx / SS, s = qidx % SS;
    int qi = randIdx[s];
    const float* qp = pc + ((size_t)b * NN + qi) * 7;
    float qx = qp[0], qy = qp[1], qz = qp[2];

    int ni = idxIn[(size_t)qidx * KK + lane];
    const float* p = pc + ((size_t)b * NN + ni) * 7;
    float px = p[0], py = p[1], pz = p[2];
    float clu[8];
    clu[0] = px - qx; clu[1] = py - qy; clu[2] = pz - qz;
    clu[3] = sqrtf(clu[0]*clu[0] + clu[1]*clu[1] + clu[2]*clu[2]);
    clu[4] = p[3]; clu[5] = p[4]; clu[6] = p[5]; clu[7] = p[6];

    // layer1: 64x8 (scale folded into w1f); lane = neighbor column.
    // Packed ds_store_b128 per 8 outputs; outer loop kept rolled to bound
    // live v8f weight rows (full unroll caused scratch spills).
#pragma unroll 1
    for (int o8 = 0; o8 < C1D; o8 += 8) {
      v8h pk;
#pragma unroll
      for (int oi = 0; oi < 8; ++oi) {
        int o = o8 + oi;
        v8f wrow = *(const v8f*)(w1f + o * 8);
        float a = be1[o];
#pragma unroll
        for (int c = 0; c < 8; ++c) a += wrow[c] * clu[c];
        a = a > 0.f ? a : 0.f;
        pk[oi] = (_Float16)a;
      }
      *(v8h*)(h1h + lane * C1D + o8) = pk;     // column-major store
    }
    // layers 2,3 on the matrix core
    wave_layer_wmma(w2h, C2D, C1D, cb2, h1h, h2h, nullptr, lane);
    wave_layer_wmma(w3h, C3D, C2D, cb3, h2h, nullptr, embf, lane);

    // attention: softmax over k of channel-max (per-lane contiguous v8f scan)
    const float* ec = embf + lane * C3D;
    v8f m8 = *(const v8f*)ec;
    for (int c = 8; c < C3D; c += 8) {
      v8f t = *(const v8f*)(ec + c);
#pragma unroll
      for (int r = 0; r < 8; ++r) m8[r] = fmaxf(m8[r], t[r]);
    }
    float am = m8[0];
#pragma unroll
    for (int r = 1; r < 8; ++r) am = fmaxf(am, m8[r]);
    float mx = am;
    for (int off = 16; off > 0; off >>= 1) mx = fmaxf(mx, __shfl_xor(mx, off, 32));
    float e  = __expf(am - mx);
    float sm = e;
    for (int off = 16; off > 0; off >>= 1) sm += __shfl_xor(sm, off, 32);
    float attn = e / sm;
    attl[lane] = attn;

    // keypoints: attention-weighted neighbor xyz
    float kx = attn * px, ky = attn * py, kz = attn * pz;
    for (int off = 16; off > 0; off >>= 1) {
      kx += __shfl_xor(kx, off, 32);
      ky += __shfl_xor(ky, off, 32);
      kz += __shfl_xor(kz, off, 32);
    }
    if (lane == 0) {
      size_t base = (size_t)b * 3 * SS + s;
      kpOut[base]          = kx;
      kpOut[base + SS]     = ky;
      kpOut[base + 2 * SS] = kz;
    }
    // pooled features: lane owns 8 contiguous channels -> v8f LDS loads,
    // one coalesced 32B global store per lane.
    v8f g8 = {};
    for (int k = 0; k < KK; ++k) {
      float a = attl[k];
      v8f ev = *(const v8f*)(embf + k * C3D + lane * 8);
#pragma unroll
      for (int r = 0; r < 8; ++r) g8[r] += ev[r] * a;
    }
    *(v8f*)(gfeatOut + (size_t)qidx * C3D + lane * 8) = g8;
  }
}

// ---------------------------------------------------------------------------
// Kernel C: head GEMM 256x256 over 8192 columns (WMMA, 2x2 register tile per
// wave -- one-shot, so no loop-carried register pressure).  BN scale folded
// into weights; bias folded into cb = scale*bias + beta.  mode 0: gfeat(f32,
// [n][256]) -> g1(f16 ws,[n][256]).  mode 1: g1 -> g2 (LDS) + fused m3 dot /
// softplus saliency.  B-tile stages are linear LDS copies.
// ---------------------------------------------------------------------------
__global__ void head_gemm(const float* __restrict__ Wf,
                          const float* __restrict__ bias,
                          const float* __restrict__ gam,
                          const float* __restrict__ bet,
                          const float* __restrict__ Bf32,
                          const _Float16* __restrict__ Bf16,
                          int mode,
                          _Float16* __restrict__ outH,
                          const float* __restrict__ m3w,
                          const float* __restrict__ m3b,
                          float* __restrict__ salOut) {
  extern __shared__ unsigned char smemC[];
  _Float16* Wh  = (_Float16*)smemC;        // 65536 halves (swizzled, scaled)
  _Float16* Bt  = Wh + 65536;              // [32][256] halves, col-major
  _Float16* g2t = Bt + 8192;               // [32][256] halves, col-major
  float* cb  = (float*)(g2t + 8192);       // 256: scale*bias + beta
  float* m3l = cb + 256;                   // 256

  int tid = threadIdx.x, lane = tid & 31, wv = tid >> 5;
  int n0  = blockIdx.x * 32;

  swizzle_weight_scaled(Wf, gam, Wh, 256, 256, tid, 256);
  if (tid < 256) {
    cb[tid]  = gam[tid] * INV_BN * bias[tid] + bet[tid];
    m3l[tid] = m3w[tid];
  }
  if (mode == 0) {
    const float* src = Bf32 + (size_t)n0 * 256;      // [n][256] f32, linear
    for (int i = tid; i < 8192; i += 256) Bt[i] = (_Float16)src[i];
  } else {
    const uint4* src = (const uint4*)(Bf16 + (size_t)n0 * 256);  // 16KB linear
    uint4* dst = (uint4*)Bt;
    for (int i = tid; i < 1024; i += 256) dst[i] = src[i];
  }
  __syncthreads();

  // each wave owns one mt-pair (rows wv*32 .. wv*32+31), full 32 columns
  int mt0 = wv * 2, mt1 = mt0 + 1;
  v8f a00 = {}, a01 = {}, a10 = {}, a11 = {};
  for (int kk = 0; kk < 8; ++kk) {
    v16h fa0 = load_frag_a_sw(Wh, 8, mt0, kk, lane);
    v16h fa1 = load_frag_a_sw(Wh, 8, mt1, kk, lane);
    v16h fb0 = load_frag_b_cm(Bt, 256, kk * 32, 0,  lane);
    v16h fb1 = load_frag_b_cm(Bt, 256, kk * 32, 16, lane);
    a00 = WMMA16(fa0, fb0, a00);
    a01 = WMMA16(fa0, fb1, a01);
    a10 = WMMA16(fa1, fb0, a10);
    a11 = WMMA16(fa1, fb1, a11);
  }
  int col0 = lane & 15;
  int r0 = mt0 * 16 + ((lane >> 4) << 3);
  int r1 = r0 + 16;
  v8f cb0 = *(const v8f*)(cb + r0);
  v8f cb1 = *(const v8f*)(cb + r1);
  v8h oh00, oh01, oh10, oh11;
#pragma unroll
  for (int r = 0; r < 8; ++r) {
    float v00 = a00[r] + cb0[r]; v00 = v00 > 0.f ? v00 : 0.f;
    float v01 = a01[r] + cb0[r]; v01 = v01 > 0.f ? v01 : 0.f;
    float v10 = a10[r] + cb1[r]; v10 = v10 > 0.f ? v10 : 0.f;
    float v11 = a11[r] + cb1[r]; v11 = v11 > 0.f ? v11 : 0.f;
    oh00[r] = (_Float16)v00;
    oh01[r] = (_Float16)v01;
    oh10[r] = (_Float16)v10;
    oh11[r] = (_Float16)v11;
  }
  if (mode == 0) {
    *(v8h*)(outH + (size_t)(n0 + col0) * 256 + r0)      = oh00;
    *(v8h*)(outH + (size_t)(n0 + col0 + 16) * 256 + r0) = oh01;
    *(v8h*)(outH + (size_t)(n0 + col0) * 256 + r1)      = oh10;
    *(v8h*)(outH + (size_t)(n0 + col0 + 16) * 256 + r1) = oh11;
  } else {
    *(v8h*)(g2t + col0 * 256 + r0)        = oh00;
    *(v8h*)(g2t + (col0 + 16) * 256 + r0) = oh01;
    *(v8h*)(g2t + col0 * 256 + r1)        = oh10;
    *(v8h*)(g2t + (col0 + 16) * 256 + r1) = oh11;
  }

  if (mode == 1) {
    __syncthreads();
    if (tid < 32) {
      float x = 0.f;
      const _Float16* gc = g2t + tid * 256;
      for (int c = 0; c < 256; ++c) x += m3l[c] * (float)gc[c];
      x += m3b[0];
      float sp = fmaxf(x, 0.f) + log1pf(__expf(-fabsf(x)));   // stable softplus
      salOut[n0 + tid] = sp + 0.001f;
    }
  }
}

// ---------------------------------------------------------------------------
extern "C" void kernel_launch(void* const* d_in, const int* in_sizes, int n_in,
                              void* d_out, int out_size, void* d_ws, size_t ws_size,
                              hipStream_t stream) {
  const float* pc      = (const float*)d_in[0];
  const int*   randIdx = (const int*)d_in[1];
  const int*   dil     = (const int*)d_in[2];
  const float* w1  = (const float*)d_in[3];
  const float* g1  = (const float*)d_in[4];
  const float* b1  = (const float*)d_in[5];
  const float* w2  = (const float*)d_in[6];
  const float* g2  = (const float*)d_in[7];
  const float* b2  = (const float*)d_in[8];
  const float* w3  = (const float*)d_in[9];
  const float* g3  = (const float*)d_in[10];
  const float* b3  = (const float*)d_in[11];
  const float* m1w = (const float*)d_in[12];
  const float* m1b = (const float*)d_in[13];
  const float* m1g = (const float*)d_in[14];
  const float* m1be= (const float*)d_in[15];
  const float* m2w = (const float*)d_in[16];
  const float* m2b = (const float*)d_in[17];
  const float* m2g = (const float*)d_in[18];
  const float* m2be= (const float*)d_in[19];
  const float* m3w = (const float*)d_in[20];
  const float* m3b = (const float*)d_in[21];

  float* kpOut  = (float*)d_out;                 // (B,3,S) = 24576 floats
  float* salOut = kpOut + BB * 3 * SS;           // (B,S)   =  8192 floats

  // workspace: [0,1MB) neighbor indices, [1MB,9MB) gfeat f32 [q][256],
  //            [9MB,13MB) g1 f16 [q][256]
  int*      idxWS   = (int*)d_ws;
  float*    gfeatWS = (float*)((char*)d_ws + (1u << 20));
  _Float16* g1WS    = (_Float16*)((char*)d_ws + (9u << 20));

  size_t ldsA = (size_t)NN * 4 + 256 * 8 + 8 + 4 * 4 + KD * 4;
  knn_topk<<<NQ, 256, ldsA, stream>>>(pc, randIdx, dil, idxWS);

  size_t ldsB = 960 * 4 + (8192 + 32768) * 2 + 4 * (size_t)PER_WAVE_BYTES;
  cluster_mlp<<<256, 128, ldsB, stream>>>(pc, randIdx, idxWS,
                                          w1, g1, b1, w2, g2, b2, w3, g3, b3,
                                          kpOut, gfeatWS);

  size_t ldsC = 65536 * 2 + 8192 * 2 + 8192 * 2 + 2 * 256 * 4;
  head_gemm<<<NQ / 32, 256, ldsC, stream>>>(m1w, m1b, m1g, m1be,
                                            gfeatWS, nullptr, 0, g1WS,
                                            m3w, m3b, salOut);
  head_gemm<<<NQ / 32, 256, ldsC, stream>>>(m2w, m2b, m2g, m2be,
                                            nullptr, g1WS, 1, nullptr,
                                            m3w, m3b, salOut);
}
